// CausalSelfAttention_70463233458826
// MI455X (gfx1250) — compile-verified, matrix operand order checked by reading, not run on hardware
//
#include <hip/hip_runtime.h>
#include <hip/hip_bf16.h>

// ---------- types ----------
typedef __attribute__((ext_vector_type(16))) __bf16 bf16x16;
typedef __attribute__((ext_vector_type(8)))  float  f32x8;

union Frag {
    bf16x16 v;
    uint4   q[2];
};

__device__ __forceinline__ unsigned short f2bf(float f) {
    unsigned int u = __float_as_uint(f);
    unsigned int r = u + 0x7FFFu + ((u >> 16) & 1u);   // round-to-nearest-even
    return (unsigned short)(r >> 16);
}

__device__ __forceinline__ f32x8 wmma_bf16(const Frag& a, const Frag& b, f32x8 c) {
    // 8 args: (neg_a, A, neg_b, B, c_mod, C, reuse_a, reuse_b)
    return __builtin_amdgcn_wmma_f32_16x16x32_bf16(false, a.v, false, b.v,
                                                   (short)0, c, false, false);
}

// ---------- CDNA5 async global->LDS path (guarded; falls back to reg staging) ----
#if defined(__AMDGCN__) && __has_builtin(__builtin_amdgcn_global_load_async_to_lds_b128)
#define HAVE_ASYNC_LDS 1
typedef int v4i_gcc __attribute__((vector_size(16)));           // matches builtin's V4i
typedef __attribute__((address_space(1))) v4i_gcc* gas_p;       // global (device) AS
typedef __attribute__((address_space(3))) v4i_gcc* las_p;       // LDS AS
__device__ __forceinline__ void async_cp16(const void* g, void* l) {
    __builtin_amdgcn_global_load_async_to_lds_b128((gas_p)g, (las_p)l, 0, 0);
}
__device__ __forceinline__ void wait_async0() {
#if __has_builtin(__builtin_amdgcn_s_wait_asynccnt)
    __builtin_amdgcn_s_wait_asynccnt(0);
#else
    asm volatile("s_wait_asynccnt 0" ::: "memory");
#endif
}
#else
#define HAVE_ASYNC_LDS 0
#endif

// ---------- 1) f32 -> bf16 elementwise (x) ----------
__global__ __launch_bounds__(256) void cvt_f32_bf16(const float* __restrict__ in,
                                                    unsigned short* __restrict__ out) {
    size_t i = ((size_t)blockIdx.x * 256 + threadIdx.x) * 8;
    float4 a = *(const float4*)(in + i);
    float4 b = *(const float4*)(in + i + 4);
    uint4 o;
    o.x = (unsigned)f2bf(a.x) | ((unsigned)f2bf(a.y) << 16);
    o.y = (unsigned)f2bf(a.z) | ((unsigned)f2bf(a.w) << 16);
    o.z = (unsigned)f2bf(b.x) | ((unsigned)f2bf(b.y) << 16);
    o.w = (unsigned)f2bf(b.z) | ((unsigned)f2bf(b.w) << 16);
    *(uint4*)(out + i) = o;
}

// ---------- 2) f32 [K][N] -> bf16 [N][K] transpose-convert (weights) ----------
__global__ __launch_bounds__(256) void transpose_f32_bf16(const float* __restrict__ W,
                                                          unsigned short* __restrict__ Wt,
                                                          int K, int N) {
    int n = blockIdx.y;
    int k = blockIdx.x * 256 + threadIdx.x;
    Wt[(size_t)n * K + k] = f2bf(W[(size_t)k * N + n]);   // coalesced stores
}

// ---------- 3) BF16 WMMA GEMM:  C[M,N](ldc) = A[M,K] * Bt[N,K]^T ----------
// Block tile 256x128, BK=32, double-buffered LDS, wave tile 64x64 (16 WMMA/K-step).
template <bool OUT_BF16>
__global__ __launch_bounds__(256) void gemm_bf16_wmma(const unsigned short* __restrict__ A,
                                                      const unsigned short* __restrict__ Bt,
                                                      void* __restrict__ C,
                                                      int M, int N, int K, int ldc) {
    constexpr int BM = 256, BN = 128, BK = 32, LD = 40;   // LD pad keeps 16B align
    __shared__ alignas(16) unsigned short As[2][BM * LD]; // 2 x 20 KiB
    __shared__ alignas(16) unsigned short Bs[2][BN * LD]; // 2 x 10 KiB

    const int tid  = threadIdx.x;
    const int lane = tid & 31;
    const int wave = tid >> 5;
    const int m0 = blockIdx.y * BM;
    const int n0 = blockIdx.x * BN;
    const int wm = (wave & 3) * 64;   // 4 wave rows (64 M each)
    const int wn = (wave >> 2) * 64;  // 2 wave cols (64 N each)
    const int hs = (lane >> 4) * 8;   // half-select per ISA A-layout
    const int lr = lane & 15;

    f32x8 acc[4][4] = {};
    const int nt = K / BK;

#if HAVE_ASYNC_LDS
    // ---- async global->LDS double-buffer pipeline (ASYNCcnt-tracked DMA) ----
    auto stage = [&](int t, int d) {
        const int k0 = t * BK;
        #pragma unroll
        for (int i = 0; i < 4; ++i) {            // A: 1024 16B chunks / 256 thr
            int c = tid + i * 256, row = c >> 2, j = (c & 3) * 8;
            async_cp16(A + (size_t)(m0 + row) * K + k0 + j, &As[d][row * LD + j]);
        }
        #pragma unroll
        for (int i = 0; i < 2; ++i) {            // B: 512 16B chunks / 256 thr
            int c = tid + i * 256, row = c >> 2, j = (c & 3) * 8;
            async_cp16(Bt + (size_t)(n0 + row) * K + k0 + j, &Bs[d][row * LD + j]);
        }
    };
    stage(0, 0);
    wait_async0();
    __syncthreads();
    for (int t = 0; t < nt; ++t) {
        const int cur = t & 1;
        if (t + 1 < nt) stage(t + 1, cur ^ 1);   // DMA next tile during compute
        Frag a[4];
        #pragma unroll
        for (int i = 0; i < 4; ++i) {
            int row = wm + i * 16 + lr;
            a[i].q[0] = *(const uint4*)(&As[cur][row * LD + hs]);
            a[i].q[1] = *(const uint4*)(&As[cur][row * LD + hs + 16]);
        }
        #pragma unroll
        for (int j = 0; j < 4; ++j) {
            Frag b;
            int row = wn + j * 16 + lr;
            b.q[0] = *(const uint4*)(&Bs[cur][row * LD + hs]);
            b.q[1] = *(const uint4*)(&Bs[cur][row * LD + hs + 16]);
            #pragma unroll
            for (int i = 0; i < 4; ++i)
                acc[i][j] = wmma_bf16(a[i], b, acc[i][j]);
        }
        wait_async0();
        __syncthreads();
    }
#else
    // ---- register-staged double-buffer pipeline (fallback) ----
    uint4 ra[4], rb[2];
    auto gload = [&](int t) {
        const int k0 = t * BK;
        #pragma unroll
        for (int i = 0; i < 4; ++i) {
            int c = tid + i * 256, row = c >> 2, j = (c & 3) * 8;
            ra[i] = *(const uint4*)(A + (size_t)(m0 + row) * K + k0 + j);
        }
        #pragma unroll
        for (int i = 0; i < 2; ++i) {
            int c = tid + i * 256, row = c >> 2, j = (c & 3) * 8;
            rb[i] = *(const uint4*)(Bt + (size_t)(n0 + row) * K + k0 + j);
        }
    };
    auto lstore = [&](int d) {
        #pragma unroll
        for (int i = 0; i < 4; ++i) {
            int c = tid + i * 256, row = c >> 2, j = (c & 3) * 8;
            *(uint4*)(&As[d][row * LD + j]) = ra[i];
        }
        #pragma unroll
        for (int i = 0; i < 2; ++i) {
            int c = tid + i * 256, row = c >> 2, j = (c & 3) * 8;
            *(uint4*)(&Bs[d][row * LD + j]) = rb[i];
        }
    };
    gload(0);
    lstore(0);
    __syncthreads();
    for (int t = 0; t < nt; ++t) {
        const int cur = t & 1;
        if (t + 1 < nt) {
            gload(t + 1);                                  // overlap with WMMA
            __builtin_prefetch(A + (size_t)(m0 + (tid >> 1)) * K + (t + 1) * BK, 0, 3);
        }
        Frag a[4];
        #pragma unroll
        for (int i = 0; i < 4; ++i) {
            int row = wm + i * 16 + lr;
            a[i].q[0] = *(const uint4*)(&As[cur][row * LD + hs]);
            a[i].q[1] = *(const uint4*)(&As[cur][row * LD + hs + 16]);
        }
        #pragma unroll
        for (int j = 0; j < 4; ++j) {
            Frag b;
            int row = wn + j * 16 + lr;
            b.q[0] = *(const uint4*)(&Bs[cur][row * LD + hs]);
            b.q[1] = *(const uint4*)(&Bs[cur][row * LD + hs + 16]);
            #pragma unroll
            for (int i = 0; i < 4; ++i)
                acc[i][j] = wmma_bf16(a[i], b, acc[i][j]);
        }
        __syncthreads();           // LDS reads done before next buffer overwrite
        if (t + 1 < nt) {
            lstore(cur ^ 1);
            __syncthreads();
        }
    }
#endif

    // C layout: row = tile_m + g + 8*(lane>=16), col = tile_n + (lane&15)
    #pragma unroll
    for (int i = 0; i < 4; ++i) {
        int rb_ = m0 + wm + i * 16 + ((lane >> 4) ? 8 : 0);
        #pragma unroll
        for (int j = 0; j < 4; ++j) {
            int col = n0 + wn + j * 16 + lr;
            #pragma unroll
            for (int g = 0; g < 8; ++g) {
                size_t idx = (size_t)(rb_ + g) * ldc + col;
                if constexpr (OUT_BF16)
                    ((unsigned short*)C)[idx] = f2bf(acc[i][j][g]);
                else
                    ((float*)C)[idx] = acc[i][j][g];
            }
        }
    }
}

// ---------- 4) causal flash attention (bf16 in/out, f32 accum) ----------
__global__ __launch_bounds__(256) void attn_flash(const unsigned short* __restrict__ qkv, // [B,T,3,H,hd]
                                                  unsigned short* __restrict__ yb,        // [B,T,H,hd]
                                                  int B_, int T, int H, int hd) {
    constexpr int BR = 64, BC = 32, HD = 128;
    constexpr int LQ = 136, LK = 136, LV = 40, LS = 68, LP = 40;
    __shared__ alignas(16) unsigned short Qs[BR * LQ];
    __shared__ alignas(16) unsigned short Ks[BC * LK];
    __shared__ alignas(16) unsigned short Vt[HD * LV];    // V transposed [d][kc]
    __shared__ alignas(16) float Sf[BR * LS];
    __shared__ float rowmax[BR], rowsum[BR], rowalpha[BR];
    unsigned short* Ps = Ks;     // P (bf16) aliases dead K tile: 64*40 <= 32*136

    const int tid = threadIdx.x, lane = tid & 31, wave = tid >> 5;
    const int qb = blockIdx.x, bh = blockIdx.y;
    const int b = bh / H, h = bh % H;
    const size_t rs = (size_t)3 * H * hd;     // qkv row stride = 6144

    // load Q tile (comp 0): 64 x 128 bf16
    {
        int r = tid >> 2, c = (tid & 3) * 32;
        const unsigned short* src = qkv + (size_t)(b * T + qb * BR + r) * rs + h * hd + c;
        #pragma unroll
        for (int u = 0; u < 4; ++u)
            *(uint4*)(Qs + r * LQ + c + u * 8) = *(const uint4*)(src + u * 8);
    }
    if (tid < BR) { rowmax[tid] = -3.0e38f; rowsum[tid] = 0.0f; }

    const int wm = wave & 3, wg = wave >> 2;   // O-phase mapping: 4 m-tiles x 2 groups
    const int sm = wave >> 1, sn = wave & 1;   // S-phase mapping: 4 m-tiles x 2 n-tiles
    const int hs = (lane >> 4) * 8;
    const int lr = lane & 15;
    const float scale = 0.08838834764831845f;  // 1/sqrt(128)
    f32x8 oacc[4] = {};

    const int kbmax = 2 * qb + 1;              // causal: skip fully-masked K blocks
    for (int kb = 0; kb <= kbmax; ++kb) {
        __syncthreads();                       // protect Ps/Vt readers of prev iter
        // stage K tile [kc][d] and V tile transposed [d][kc]
        {
            int kc = tid >> 3, c = (tid & 7) * 16;
            const unsigned short* kr =
                qkv + (size_t)(b * T + kb * BC + kc) * rs + (size_t)H * hd + h * hd + c;
            *(uint4*)(Ks + kc * LK + c)     = *(const uint4*)(kr);
            *(uint4*)(Ks + kc * LK + c + 8) = *(const uint4*)(kr + 8);
            const unsigned short* vr =
                qkv + (size_t)(b * T + kb * BC + kc) * rs + (size_t)2 * H * hd + h * hd + c;
            unsigned short tmp[16];
            *(uint4*)(tmp)     = *(const uint4*)(vr);
            *(uint4*)(tmp + 8) = *(const uint4*)(vr + 8);
            #pragma unroll
            for (int u = 0; u < 16; ++u) Vt[(c + u) * LV + kc] = tmp[u];
        }
        __syncthreads();

        // S = Q * K^T : one 16x16 tile per wave, K-dim = hd = 4 x 32
        {
            f32x8 s = {};
            #pragma unroll
            for (int ks = 0; ks < 4; ++ks) {
                Frag a, bk;
                int ar = sm * 16 + lr;
                a.q[0] = *(const uint4*)(Qs + ar * LQ + ks * 32 + hs);
                a.q[1] = *(const uint4*)(Qs + ar * LQ + ks * 32 + hs + 16);
                int br = sn * 16 + lr;
                bk.q[0] = *(const uint4*)(Ks + br * LK + ks * 32 + hs);
                bk.q[1] = *(const uint4*)(Ks + br * LK + ks * 32 + hs + 16);
                s = wmma_bf16(a, bk, s);
            }
            int col = sn * 16 + lr;
            int rb  = sm * 16 + ((lane >> 4) ? 8 : 0);
            #pragma unroll
            for (int g = 0; g < 8; ++g) Sf[(rb + g) * LS + col] = s[g];
        }
        __syncthreads();

        // online softmax, one row per thread
        if (tid < BR) {
            int r = tid, qpos = qb * BR + r;
            float mprev = rowmax[r], bmax = -3.0e38f;
            for (int c = 0; c < BC; ++c) {
                float sv = Sf[r * LS + c] * scale;
                if (kb * BC + c > qpos) sv = -3.0e38f;
                Sf[r * LS + c] = sv;
                bmax = fmaxf(bmax, sv);
            }
            float nm = fmaxf(mprev, bmax);
            float al = __expf(mprev - nm);
            float ls = 0.0f;
            for (int c = 0; c < BC; ++c) {
                float p = __expf(Sf[r * LS + c] - nm);
                ls += p;
                Ps[r * LP + c] = f2bf(p);
            }
            rowmax[r]   = nm;
            rowsum[r]   = rowsum[r] * al + ls;
            rowalpha[r] = al;
        }
        __syncthreads();

        // O = O*alpha + P*V : 4 n-tiles per wave, K-dim = BC = 32 (one WMMA each)
        {
            float al8[8];
            int rb = wm * 16 + ((lane >> 4) ? 8 : 0);
            #pragma unroll
            for (int g = 0; g < 8; ++g) al8[g] = rowalpha[rb + g];
            #pragma unroll
            for (int j = 0; j < 4; ++j)
                #pragma unroll
                for (int g = 0; g < 8; ++g) oacc[j][g] *= al8[g];

            Frag a;
            int ar = wm * 16 + lr;
            a.q[0] = *(const uint4*)(Ps + ar * LP + hs);
            a.q[1] = *(const uint4*)(Ps + ar * LP + hs + 16);
            #pragma unroll
            for (int j = 0; j < 4; ++j) {
                Frag bv;
                int br = (wg * 4 + j) * 16 + lr;
                bv.q[0] = *(const uint4*)(Vt + br * LV + hs);
                bv.q[1] = *(const uint4*)(Vt + br * LV + hs + 16);
                oacc[j] = wmma_bf16(a, bv, oacc[j]);
            }
        }
    }

    __syncthreads();
    if (tid < BR) rowalpha[tid] = 1.0f / rowsum[tid];
    __syncthreads();

    // write O (bf16) to y: [B,T,H,hd]
    {
        int rb = wm * 16 + ((lane >> 4) ? 8 : 0);
        #pragma unroll
        for (int j = 0; j < 4; ++j) {
            int col = (wg * 4 + j) * 16 + lr;
            #pragma unroll
            for (int g = 0; g < 8; ++g) {
                int row = rb + g;
                float v = oacc[j][g] * rowalpha[row];
                yb[(size_t)(b * T + qb * BR + row) * (H * hd) + h * hd + col] = f2bf(v);
            }
        }
    }
}

// ---------- launcher ----------
extern "C" void kernel_launch(void* const* d_in, const int* in_sizes, int n_in,
                              void* d_out, int out_size, void* d_ws, size_t ws_size,
                              hipStream_t stream) {
    (void)in_sizes; (void)n_in; (void)out_size; (void)ws_size;
    const float* x     = (const float*)d_in[0];   // [4,2048,2048]
    const float* Wqkv  = (const float*)d_in[1];   // [2048,6144]
    const float* Wproj = (const float*)d_in[2];   // [2048,2048]
    float* out = (float*)d_out;                   // [4,2048,2048] f32

    const int B = 4, T = 2048, C = 2048, H = 16, hd = 128;
    const int M = B * T;        // 8192
    const int N1 = 3 * C;       // 6144

    char* ws = (char*)d_ws;
    unsigned short* xb     = (unsigned short*)(ws);                 // 32 MiB
    unsigned short* wqkvT  = (unsigned short*)(ws + 33554432ull);   // 24 MiB
    unsigned short* wprojT = (unsigned short*)(ws + 58720256ull);   //  8 MiB
    unsigned short* qkvb   = (unsigned short*)(ws + 67108864ull);   // 96 MiB
    unsigned short* ybuf   = (unsigned short*)(ws + 167772160ull);  // 32 MiB

    // 1) precision conversion / weight transposition (one-time, bandwidth-trivial)
    cvt_f32_bf16<<<(size_t)M * C / 2048, 256, 0, stream>>>(x, xb);
    transpose_f32_bf16<<<dim3(C / 256, N1), 256, 0, stream>>>(Wqkv, wqkvT, C, N1);
    transpose_f32_bf16<<<dim3(C / 256, C), 256, 0, stream>>>(Wproj, wprojT, C, C);

    // 2) QKV projection (bf16 WMMA, bf16 out)
    gemm_bf16_wmma<true><<<dim3(N1 / 128, M / 256), 256, 0, stream>>>(
        xb, wqkvT, qkvb, M, N1, C, N1);

    // 3) causal flash attention (bf16 WMMA, f32 online softmax)
    attn_flash<<<dim3(T / 64, B * H), 256, 0, stream>>>(qkvb, ybuf, B, T, H, hd);

    // 4) output projection (bf16 WMMA, f32 out)
    gemm_bf16_wmma<false><<<dim3(C / 128, M / 256), 256, 0, stream>>>(
        ybuf, wprojT, out, M, C, C, C);
}
